// SEGNN_Layer_55267639165390
// MI455X (gfx1250) — matrix operand order repaired and checked by reading, not compiled, not run on hardware
//
#include <hip/hip_runtime.h>
#include <math.h>

#define DD 128
#define EPS 1e-5f

typedef __attribute__((ext_vector_type(16))) _Float16 v16h;
typedef __attribute__((ext_vector_type(8)))  float    v8f;

union AFrag { v16h v; _Float16 h[16]; };
union CFrag { v8f  v; float    f[8];  };

// ---------------- helpers: monotonic uint key for float atomic-max -------------
__device__ __forceinline__ unsigned fkey(float f) {
    unsigned u = __float_as_uint(f);
    return (u & 0x80000000u) ? ~u : (u | 0x80000000u);
}
__device__ __forceinline__ float keyToF(unsigned k) {
    unsigned u = (k & 0x80000000u) ? (k ^ 0x80000000u) : ~k;
    return __uint_as_float(u);
}

// CDNA5 async Global->LDS copy, 16B per lane, tracked by ASYNCcnt
__device__ __forceinline__ void asyncCopy16(unsigned ldsOff, const void* g) {
    asm volatile("global_load_async_to_lds_b128 %0, %1, off"
                 :: "v"(ldsOff), "v"((unsigned long long)(size_t)g) : "memory");
}
__device__ __forceinline__ void asyncWait0() {
    asm volatile("s_wait_asynccnt 0x0" ::: "memory");
}

// ---------------- fill --------------------------------------------------------
__global__ void k_fill_u32(unsigned* p, unsigned val, long n) {
    long i = (long)blockIdx.x * blockDim.x + threadIdx.x;
    long stride = (long)gridDim.x * blockDim.x;
    for (; i < n; i += stride) p[i] = val;
}

// ---------------- f32 -> f16 weight conversion --------------------------------
__global__ void k_cvt_f16(const float* __restrict__ s, _Float16* __restrict__ d, int n) {
    int i = blockIdx.x * blockDim.x + threadIdx.x;
    if (i < n) d[i] = (_Float16)s[i];
}

// ---------------- pass 1: three attention logits + segment max ----------------
__global__ void k_logits(const float* __restrict__ ent, const float* __restrict__ rel,
                         const int* __restrict__ srcI, const int* __restrict__ dstI,
                         const int* __restrict__ etI,
                         float* __restrict__ logit, unsigned* __restrict__ maxv,
                         int E, int N) {
    int e = blockIdx.x * blockDim.x + threadIdx.x;
    if (e >= E) return;
    const float* ps = ent + (long)srcI[e] * DD;
    const float* pd = ent + (long)dstI[e] * DD;
    const float* pr = rel + (long)etI[e] * DD;
    float l0 = 0.f, l1 = 0.f, l2 = 0.f;
    for (int k = 0; k < DD; k += 4) {
        float4 s = *(const float4*)(ps + k);
        float4 d = *(const float4*)(pd + k);
        float4 r = *(const float4*)(pr + k);
        l0 += r.x*d.x + r.y*d.y + r.z*d.z + r.w*d.w;
        l1 += s.x*d.x + s.y*d.y + s.z*d.z + s.w*d.w;
        l2 += s.x*r.x*d.x + s.y*r.y*d.y + s.z*r.z*d.z + s.w*r.w*d.w;
    }
    logit[e]            = l0;
    logit[(long)E + e]  = l1;
    logit[2L*E + e]     = l2;
    int dv = dstI[e];
    atomicMax(&maxv[dv],        fkey(l0));
    atomicMax(&maxv[N + dv],    fkey(l1));
    atomicMax(&maxv[2*N + dv],  fkey(l2));
}

// ---------------- pass 2: exp + denominator ----------------------------------
__global__ void k_expdenom(const int* __restrict__ dstI, const unsigned* __restrict__ maxv,
                           float* __restrict__ logit, float* __restrict__ denom,
                           int E, int N) {
    long i = (long)blockIdx.x * blockDim.x + threadIdx.x;
    if (i >= 3L * E) return;
    int l = (int)(i / E);
    int e = (int)(i - (long)l * E);
    int dv = dstI[e];
    float m = keyToF(maxv[(long)l * N + dv]);
    float w = __expf(logit[i] - m);
    logit[i] = w;
    atomicAdd(&denom[(long)l * N + dv], w);
}

// ---------------- pass 3: WMMA per-edge transform + scatter -------------------
// One wave handles a 16-edge tile; 8 N-tiles x 4 K-steps x 2 weights = 64 WMMA.
// Weights are async-staged into LDS once per block (XOR-swizzled 32B chunks).
__global__ __launch_bounds__(256)
void k_edge_gemm(int layer,
                 const float* __restrict__ ent, const float* __restrict__ rel,
                 const int* __restrict__ srcI, const int* __restrict__ dstI,
                 const int* __restrict__ etI, const unsigned char* __restrict__ inm,
                 const _Float16* __restrict__ Wi, const _Float16* __restrict__ Wo,
                 const float* __restrict__ bi, const float* __restrict__ bo,
                 const float* __restrict__ wAttn, const float* __restrict__ denom,
                 float* __restrict__ acc, int E) {
    __shared__ _Float16 wlds[2][DD * DD];      // 64 KB: swizzled f16 Wi, Wo
    __shared__ _Float16 xs[8][16 * DD];        // 32 KB: per-wave 16x128 feature tile
    __shared__ int   dsts[8][16];
    __shared__ float aws[8][16];
    __shared__ int   msks[8][16];

    int wave = threadIdx.x >> 5;
    int lane = threadIdx.x & 31;
    long tile  = (long)blockIdx.x * 8 + wave;
    long eBase = tile * 16;

    // --- async-stage both weight matrices into LDS (16B units, swizzled) ------
    {
        const _Float16* wsrc0 = Wi;
        const _Float16* wsrc1 = Wo;
        for (int t = threadIdx.x; t < 2 * DD * 16; t += 256) {
            int mtx = t >> 11;            // matrix 0/1
            int row = (t >> 4) & 127;     // output column (row of W)
            int sub = t & 15;             // 16B unit within the 256B row
            int phys = (((sub >> 1) ^ (row & 7)) << 1) | (sub & 1);
            unsigned ldsOff = (unsigned)(size_t)&wlds[mtx][row * DD + phys * 8];
            const _Float16* g = (mtx ? wsrc1 : wsrc0) + (size_t)row * DD + sub * 8;
            asyncCopy16(ldsOff, g);
        }
        asyncWait0();
    }

    // --- stage per-edge scalars ----------------------------------------------
    if (lane < 16) {
        long e = eBase + lane;
        if (e < E) {
            int dv = dstI[e];
            dsts[wave][lane] = dv;
            aws[wave][lane]  = wAttn[e] / denom[dv];
            msks[wave][lane] = inm[e] ? 1 : 0;
        } else {
            dsts[wave][lane] = -1;
            aws[wave][lane]  = 0.f;
            msks[wave][lane] = 0;
        }
    }
    // --- stage 16x128 feature tile (f16) -------------------------------------
    for (int t = lane; t < 16 * DD; t += 32) {
        int m = t >> 7, k = t & 127;
        long e = eBase + m;
        float x = 0.f;
        if (e < E) {
            if (layer == 0)      x = rel[(long)etI[e] * DD + k];
            else if (layer == 1) x = ent[(long)srcI[e] * DD + k];
            else                 x = ent[(long)srcI[e] * DD + k] * rel[(long)etI[e] * DD + k];
        }
        xs[wave][t] = (_Float16)x;
    }
    __syncthreads();

    int khalf = lane >> 4;   // 0: lanes 0-15, 1: lanes 16-31
    int ln16  = lane & 15;

    for (int nb = 0; nb < 8; ++nb) {
        int col = nb * 16 + ln16;

        CFrag ci, co;
        #pragma unroll
        for (int r = 0; r < 8; ++r) { ci.f[r] = 0.f; co.f[r] = 0.f; }

        for (int kb = 0; kb < 4; ++kb) {
            // A fragment: 16x32 f16; lane holds row M=ln16, K-half by khalf
            AFrag a;
            #pragma unroll
            for (int v = 0; v < 8; ++v) {
                int kbase = 2*v + khalf*16;
                int K = kb * 32 + kbase;
                a.h[2*v]   = xs[wave][ln16 * DD + K];
                a.h[2*v+1] = xs[wave][ln16 * DD + K + 1];
            }
            // B fragments from swizzled LDS: B[k][n] = W[n][k]
            int physc = ((kb * 2 + khalf) ^ (col & 7)) * 16;
            AFrag fbi, fbo;
            #pragma unroll
            for (int v = 0; v < 8; ++v) {
                fbi.h[2*v]   = wlds[0][col * DD + physc + 2*v];
                fbi.h[2*v+1] = wlds[0][col * DD + physc + 2*v + 1];
                fbo.h[2*v]   = wlds[1][col * DD + physc + 2*v];
                fbo.h[2*v+1] = wlds[1][col * DD + physc + 2*v + 1];
            }
            ci.v = __builtin_amdgcn_wmma_f32_16x16x32_f16(false, a.v, false, fbi.v,
                                                          (short)0, ci.v, false, false);
            co.v = __builtin_amdgcn_wmma_f32_16x16x32_f16(false, a.v, false, fbo.v,
                                                          (short)0, co.v, false, false);
        }
        float biasI = bi[col], biasO = bo[col];
        #pragma unroll
        for (int r = 0; r < 8; ++r) {
            int m  = khalf * 8 + r;      // D layout: VGPR r holds M = r (+8 for high lanes)
            int dv = dsts[wave][m];
            if (dv < 0) continue;
            float v = msks[wave][m] ? (ci.f[r] + biasI) : (co.f[r] + biasO);
            v *= aws[wave][m];
            atomicAdd(&acc[(long)dv * DD + col], v);
        }
    }
}

// ---------------- pass 4: BN column stats (coalesced) -------------------------
__global__ void k_bnstats(const float* __restrict__ acc,
                          float* __restrict__ colsum, float* __restrict__ colsq, int N) {
    int l = blockIdx.y;
    int d = threadIdx.x;               // 128 threads = 128 columns
    long base = (long)l * N * DD;
    float s = 0.f, q = 0.f;
    for (long i = blockIdx.x; i < N; i += gridDim.x) {
        float v = acc[base + i * DD + d];
        s += v; q += v * v;
    }
    atomicAdd(&colsum[l * DD + d], s);
    atomicAdd(&colsq[l * DD + d], q);
}

// ---------------- pass 5: final entity output ---------------------------------
__global__ void k_final_ent(const float* __restrict__ ent, const float* __restrict__ acc,
                            const float* __restrict__ colsum, const float* __restrict__ colsq,
                            const float* __restrict__ g0, const float* __restrict__ b0,
                            const float* __restrict__ g1, const float* __restrict__ b1,
                            const float* __restrict__ g2, const float* __restrict__ b2,
                            float* __restrict__ out, int N) {
    long i = (long)blockIdx.x * blockDim.x + threadIdx.x;
    if (i >= (long)N * DD) return;
    int dc = (int)(i & (DD - 1));
    const float* gs[3] = { g0, g1, g2 };
    const float* bs[3] = { b0, b1, b2 };
    float r = ent[i];
    float invN = 1.f / (float)N;
    #pragma unroll
    for (int l = 0; l < 3; ++l) {
        float mu  = colsum[l * DD + dc] * invN;
        float var = colsq[l * DD + dc] * invN - mu * mu;
        float v   = acc[(long)l * N * DD + i];
        r += tanhf(gs[l][dc] * (v - mu) * rsqrtf(var + EPS) + bs[l][dc]);
    }
    out[i] = r;
}

// ---------------- relation update ---------------------------------------------
__global__ void k_relcnt(const int* __restrict__ relIds, float* __restrict__ cnt, int P) {
    int p = blockIdx.x * blockDim.x + threadIdx.x;
    if (p < P) atomicAdd(&cnt[relIds[p]], 1.f);
}

__global__ __launch_bounds__(256)
void k_pair_gemm(const float* __restrict__ ent,
                 const int* __restrict__ heads, const int* __restrict__ tails,
                 const int* __restrict__ relIds,
                 const _Float16* __restrict__ Wr, const float* __restrict__ relb,
                 float* __restrict__ relacc, int P) {
    __shared__ _Float16 wlds[DD * DD];         // 32 KB swizzled relW
    __shared__ _Float16 xs[8][16 * DD];        // 32 KB
    __shared__ int rids[8][16];

    int wave = threadIdx.x >> 5;
    int lane = threadIdx.x & 31;
    long tile  = (long)blockIdx.x * 8 + wave;
    long pBase = tile * 16;

    // async-stage relW into LDS
    for (int t = threadIdx.x; t < DD * 16; t += 256) {
        int row = (t >> 4) & 127;
        int sub = t & 15;
        int phys = (((sub >> 1) ^ (row & 7)) << 1) | (sub & 1);
        unsigned ldsOff = (unsigned)(size_t)&wlds[row * DD + phys * 8];
        asyncCopy16(ldsOff, Wr + (size_t)row * DD + sub * 8);
    }
    asyncWait0();

    if (lane < 16) {
        long p = pBase + lane;
        rids[wave][lane] = (p < P) ? relIds[p] : -1;
    }
    for (int t = lane; t < 16 * DD; t += 32) {
        int m = t >> 7, k = t & 127;
        long p = pBase + m;
        float x = 0.f;
        if (p < P)
            x = ent[(long)tails[p] * DD + k] - ent[(long)heads[p] * DD + k];
        xs[wave][t] = (_Float16)x;
    }
    __syncthreads();

    int khalf = lane >> 4;
    int ln16  = lane & 15;

    for (int nb = 0; nb < 8; ++nb) {
        int col = nb * 16 + ln16;

        CFrag c;
        #pragma unroll
        for (int r = 0; r < 8; ++r) c.f[r] = 0.f;

        for (int kb = 0; kb < 4; ++kb) {
            AFrag a;
            #pragma unroll
            for (int v = 0; v < 8; ++v) {
                int kbase = 2*v + khalf*16;
                int K = kb * 32 + kbase;
                a.h[2*v]   = xs[wave][ln16 * DD + K];
                a.h[2*v+1] = xs[wave][ln16 * DD + K + 1];
            }
            int physc = ((kb * 2 + khalf) ^ (col & 7)) * 16;
            AFrag fb;
            #pragma unroll
            for (int v = 0; v < 8; ++v) {
                fb.h[2*v]   = wlds[col * DD + physc + 2*v];
                fb.h[2*v+1] = wlds[col * DD + physc + 2*v + 1];
            }
            c.v = __builtin_amdgcn_wmma_f32_16x16x32_f16(false, a.v, false, fb.v,
                                                         (short)0, c.v, false, false);
        }
        float bias = relb[col];
        #pragma unroll
        for (int r = 0; r < 8; ++r) {
            int m   = khalf * 8 + r;
            int rid = rids[wave][m];
            if (rid < 0) continue;
            atomicAdd(&relacc[(long)rid * DD + col], c.f[r] + bias);
        }
    }
}

__global__ void k_final_rel(const float* __restrict__ relEmb,
                            const float* __restrict__ relacc, const float* __restrict__ cnt,
                            float* __restrict__ out, int R) {
    long i = (long)blockIdx.x * blockDim.x + threadIdx.x;
    if (i >= (long)R * DD) return;
    int r = (int)(i >> 7);
    out[i] = relEmb[i] + tanhf(relacc[i] / fmaxf(cnt[r], 1.f));
}

// ===============================================================================
extern "C" void kernel_launch(void* const* d_in, const int* in_sizes, int n_in,
                              void* d_out, int out_size, void* d_ws, size_t ws_size,
                              hipStream_t stream) {
    const float* ent     = (const float*)d_in[0];
    const float* relEmb  = (const float*)d_in[1];
    const float* edge_Wo = (const float*)d_in[2];
    const float* edge_bo = (const float*)d_in[3];
    const float* edge_Wi = (const float*)d_in[4];
    const float* edge_bi = (const float*)d_in[5];
    const float* edge_g  = (const float*)d_in[6];
    const float* edge_b  = (const float*)d_in[7];
    const float* node_Wo = (const float*)d_in[8];
    const float* node_bo = (const float*)d_in[9];
    const float* node_Wi = (const float*)d_in[10];
    const float* node_bi = (const float*)d_in[11];
    const float* node_g  = (const float*)d_in[12];
    const float* node_b  = (const float*)d_in[13];
    const float* comp_Wo = (const float*)d_in[14];
    const float* comp_bo = (const float*)d_in[15];
    const float* comp_Wi = (const float*)d_in[16];
    const float* comp_bi = (const float*)d_in[17];
    const float* comp_g  = (const float*)d_in[18];
    const float* comp_b  = (const float*)d_in[19];
    const float* relW    = (const float*)d_in[20];
    const float* relb    = (const float*)d_in[21];
    const int* srcI   = (const int*)d_in[22];
    const int* dstI   = (const int*)d_in[23];
    const int* etI    = (const int*)d_in[24];
    const int* heads  = (const int*)d_in[25];
    const int* tails  = (const int*)d_in[26];
    const int* relIds = (const int*)d_in[27];
    const unsigned char* inm = (const unsigned char*)d_in[28];

    const int N = in_sizes[0] / DD;
    const int R = in_sizes[1] / DD;
    const int E = in_sizes[22];
    const int P = in_sizes[25];

    char* ws = (char*)d_ws;
    size_t off = 0;
    auto alloc = [&](size_t bytes) { size_t o = off; off += (bytes + 255) & ~(size_t)255; return o; };

    // f16 weights: edgeWi, edgeWo, nodeWi, nodeWo, compWi, compWo, relW
    size_t whOff[7];
    for (int i = 0; i < 7; ++i) whOff[i] = alloc((size_t)DD * DD * sizeof(_Float16));
    size_t logitOff = alloc(3ull * E * sizeof(float));
    size_t maxvOff  = alloc(3ull * N * sizeof(unsigned));
    size_t zeroOff  = off;                               // zero-region start
    size_t denomOff = alloc(3ull * N * sizeof(float));
    size_t accOff   = alloc(3ull * N * DD * sizeof(float));
    size_t csumOff  = alloc(3ull * DD * sizeof(float));
    size_t csqOff   = alloc(3ull * DD * sizeof(float));
    size_t raccOff  = alloc((size_t)R * DD * sizeof(float));
    size_t rcntOff  = alloc((size_t)R * sizeof(float));
    size_t zeroBytes = off - zeroOff;

    _Float16* whEdgeI = (_Float16*)(ws + whOff[0]);
    _Float16* whEdgeO = (_Float16*)(ws + whOff[1]);
    _Float16* whNodeI = (_Float16*)(ws + whOff[2]);
    _Float16* whNodeO = (_Float16*)(ws + whOff[3]);
    _Float16* whCompI = (_Float16*)(ws + whOff[4]);
    _Float16* whCompO = (_Float16*)(ws + whOff[5]);
    _Float16* whRel   = (_Float16*)(ws + whOff[6]);
    float*    logit   = (float*)(ws + logitOff);
    unsigned* maxv    = (unsigned*)(ws + maxvOff);
    float*    denom   = (float*)(ws + denomOff);
    float*    acc     = (float*)(ws + accOff);
    float*    colsum  = (float*)(ws + csumOff);
    float*    colsq   = (float*)(ws + csqOff);
    float*    relacc  = (float*)(ws + raccOff);
    float*    relcnt  = (float*)(ws + rcntOff);

    // -- init: maxv = key(-inf) = 0x007FFFFF, zero region = 0
    k_fill_u32<<<1024, 256, 0, stream>>>(maxv, 0x007FFFFFu, 3L * N);
    k_fill_u32<<<4096, 256, 0, stream>>>((unsigned*)(ws + zeroOff), 0u, (long)(zeroBytes / 4));

    // -- weight conversion
    const int WN = DD * DD, WB = (WN + 255) / 256;
    k_cvt_f16<<<WB, 256, 0, stream>>>(edge_Wi, whEdgeI, WN);
    k_cvt_f16<<<WB, 256, 0, stream>>>(edge_Wo, whEdgeO, WN);
    k_cvt_f16<<<WB, 256, 0, stream>>>(node_Wi, whNodeI, WN);
    k_cvt_f16<<<WB, 256, 0, stream>>>(node_Wo, whNodeO, WN);
    k_cvt_f16<<<WB, 256, 0, stream>>>(comp_Wi, whCompI, WN);
    k_cvt_f16<<<WB, 256, 0, stream>>>(comp_Wo, whCompO, WN);
    k_cvt_f16<<<WB, 256, 0, stream>>>(relW,    whRel,   WN);

    // -- attention softmax
    k_logits<<<(E + 255) / 256, 256, 0, stream>>>(ent, relEmb, srcI, dstI, etI,
                                                  logit, maxv, E, N);
    k_expdenom<<<(int)((3L * E + 255) / 256), 256, 0, stream>>>(dstI, maxv, logit, denom, E, N);

    // -- WMMA edge transforms (3 layers)
    int tiles  = (E + 15) / 16;
    int blocks = (tiles + 7) / 8;
    k_edge_gemm<<<blocks, 256, 0, stream>>>(0, ent, relEmb, srcI, dstI, etI, inm,
        whEdgeI, whEdgeO, edge_bi, edge_bo, logit, denom, acc, E);
    k_edge_gemm<<<blocks, 256, 0, stream>>>(1, ent, relEmb, srcI, dstI, etI, inm,
        whNodeI, whNodeO, node_bi, node_bo, logit + (long)E, denom + (long)N,
        acc + (long)N * DD, E);
    k_edge_gemm<<<blocks, 256, 0, stream>>>(2, ent, relEmb, srcI, dstI, etI, inm,
        whCompI, whCompO, comp_bi, comp_bo, logit + 2L * E, denom + 2L * N,
        acc + 2L * N * DD, E);

    // -- BN stats + final entity output
    k_bnstats<<<dim3(256, 3), 128, 0, stream>>>(acc, colsum, colsq, N);
    k_final_ent<<<(int)(((long)N * DD + 255) / 256), 256, 0, stream>>>(
        ent, acc, colsum, colsq,
        edge_g, edge_b, node_g, node_b, comp_g, comp_b,
        (float*)d_out, N);

    // -- relation update
    k_relcnt<<<(P + 255) / 256, 256, 0, stream>>>(relIds, relcnt, P);
    int ptiles  = (P + 15) / 16;
    int pblocks = (ptiles + 7) / 8;
    k_pair_gemm<<<pblocks, 256, 0, stream>>>(ent, heads, tails, relIds, whRel, relb, relacc, P);
    k_final_rel<<<(int)(((long)R * DD + 255) / 256), 256, 0, stream>>>(
        relEmb, relacc, relcnt, (float*)d_out + (long)N * DD, R);
}
